// MGFIN_50818053046339
// MI455X (gfx1250) — matrix operand order repaired and checked by reading, not compiled, not run on hardware
//
#include <hip/hip_runtime.h>
#include <hip/hip_bf16.h>
#include <math.h>
#include <cstddef>

// ---------------------------------------------------------------------------
// Sizes (reference: B=256, P=196, DIM=768, NP_PARTS=312, NCLS=200, W2V=300)
// Padded: PP=256 (196 -> mult of 64), LP=320 (312 -> mult of 64)
// All GEMM M/N dims are multiples of 64; all K dims multiples of 32.
// ---------------------------------------------------------------------------
constexpr int BB  = 256;
constexpr int P   = 196;
constexpr int PP  = 256;
constexpr int L   = 312;
constexpr int LP  = 320;
constexpr int D   = 768;
constexpr int HID = 3072;
constexpr int NC  = 200;
constexpr int W2VD = 300;
constexpr float ATT_SCALE = 0.03608439182435161f;   // 768^-0.5

typedef __attribute__((ext_vector_type(16))) _Float16 v16h;
typedef __attribute__((ext_vector_type(8)))  _Float16 v8h;
typedef __attribute__((ext_vector_type(8)))  float    v8f;

__device__ __forceinline__ float gelu_exact(float x) {
    return 0.5f * x * (1.0f + erff(x * 0.70710678118654752f));
}

// ---------------------------------------------------------------------------
// Register-blocked WMMA GEMM:  C[b] = act(alpha * A[b] @ W[b]^T + bias)(+resid)
//   A: f16 [Mpad, K] row-major (ld = K), per-batch stride sA
//   W: f16 [N, K]    row-major (ld = K), per-batch stride sB (0 = shared)
//   Block = 128 threads (4 waves) -> 64x64 C-tile.
//   Each wave: 16(M) x 64(N); one A-fragment feeds 4 WMMAs per K-step
//   (4x A-reuse; ~12.8 flop/B of load traffic, activations stay L2-resident).
//   Fragment layouts used here (A and B share one k order):
//     A 16x32 f16: lane m=l%16; elems 0..15 -> k = (l>=16?16:0)+0..15
//     B 32x16 f16: lane n=l%16; elems 0..15 -> k = (l>=16?16:0)+0..15
//     D f32: elem r, lane l -> m = r+(l>=16?8:0), n = l%16
// ---------------------------------------------------------------------------
__global__ __launch_bounds__(128) void wmma_gemm(
    const _Float16* __restrict__ A, long long sA,
    const _Float16* __restrict__ Bm, long long sB,
    const float* __restrict__ bias,
    const float* __restrict__ resid, long long sR, int ldR, int Mreal,
    float* __restrict__ outF, long long sOF,
    _Float16* __restrict__ outH, long long sOH,
    int N, int K, float alpha, int act, int transOut)
{
    const int lane = threadIdx.x & 31;
    const int wv   = threadIdx.x >> 5;
    const int n0 = blockIdx.x << 6;
    const int m0 = (blockIdx.y << 6) + (wv << 4);
    const int b  = blockIdx.z;

    const _Float16* arow = A  + (long long)b * sA + (long long)(m0 + (lane & 15)) * K;
    const _Float16* brow = Bm + (long long)b * sB + (long long)(n0 + (lane & 15)) * K;
    const int aoff = (lane & 16) ? 8 : 0;
    const int boff = (lane & 16) ? 16 : 0;
    const long long btile = (long long)16 * K;   // stride between B n-tiles

    v8f acc0 = {}, acc1 = {}, acc2 = {}, acc3 = {};
    for (int k0 = 0; k0 < K; k0 += 32) {
        v8h  alo = *(const v8h*)(arow + k0 + boff);
        v8h  ahi = *(const v8h*)(arow + k0 + boff + 8);
        v16h af  = __builtin_shufflevector(alo, ahi,
                     0,1,2,3,4,5,6,7,8,9,10,11,12,13,14,15);
        v16h b0 = *(const v16h*)(brow              + k0 + boff);
        v16h b1 = *(const v16h*)(brow + btile      + k0 + boff);
        v16h b2 = *(const v16h*)(brow + 2 * btile  + k0 + boff);
        v16h b3 = *(const v16h*)(brow + 3 * btile  + k0 + boff);
        acc0 = __builtin_amdgcn_wmma_f32_16x16x32_f16(false, af, false, b0, (short)0, acc0, false, false);
        acc1 = __builtin_amdgcn_wmma_f32_16x16x32_f16(false, af, false, b1, (short)0, acc1, false, false);
        acc2 = __builtin_amdgcn_wmma_f32_16x16x32_f16(false, af, false, b2, (short)0, acc2, false, false);
        acc3 = __builtin_amdgcn_wmma_f32_16x16x32_f16(false, af, false, b3, (short)0, acc3, false, false);
    }

    const int mb   = m0 + ((lane & 16) ? 8 : 0);
    const int Mpad = (int)(gridDim.y << 6);

#pragma unroll
    for (int j = 0; j < 4; ++j) {
        const v8f acc = (j == 0) ? acc0 : (j == 1) ? acc1 : (j == 2) ? acc2 : acc3;
        const int n  = n0 + (j << 4) + (lane & 15);
        const float bv = bias ? bias[n] : 0.0f;
#pragma unroll
        for (int r = 0; r < 8; ++r) {
            int m = mb + r;
            float x = acc[r] * alpha + bv;
            if (act == 1)      x = fmaxf(x, 0.0f);
            else if (act == 2) x = gelu_exact(x);
            if (resid && m < Mreal)
                x += resid[(long long)b * sR + (long long)m * ldR + n];
            long long idx = transOut ? ((long long)n * Mpad + m)
                                     : ((long long)m * N + n);
            if (outF) outF[(long long)b * sOF + idx] = x;
            if (outH) outH[(long long)b * sOH + idx] = (_Float16)x;
        }
    }
}

// ---------------------------------------------------------------------------
// LayerNorm over last dim; writes up to 3 (w,b) variants as f16.
// Pad rows (row >= Mreal) are zero-filled (keeps WMMA operands clean).
// ---------------------------------------------------------------------------
__global__ __launch_bounds__(128) void ln_kernel(
    const float* __restrict__ x, long long sX, int ldX, int Mreal, int Dd,
    long long sO,
    const float* __restrict__ w0, const float* __restrict__ b0, _Float16* __restrict__ o0,
    const float* __restrict__ w1, const float* __restrict__ b1, _Float16* __restrict__ o1,
    const float* __restrict__ w2, const float* __restrict__ b2, _Float16* __restrict__ o2)
{
    const int row = blockIdx.x, b = blockIdx.y, t = threadIdx.x;
    __shared__ float red[128];
    const long long ob = (long long)b * sO + (long long)row * Dd;

    if (row >= Mreal) {
        for (int d = t; d < Dd; d += 128) {
            if (o0) o0[ob + d] = (_Float16)0.0f;
            if (o1) o1[ob + d] = (_Float16)0.0f;
            if (o2) o2[ob + d] = (_Float16)0.0f;
        }
        return;
    }
    const float* xr = x + (long long)b * sX + (long long)row * ldX;

    float s = 0.0f;
    for (int d = t; d < Dd; d += 128) s += xr[d];
    red[t] = s; __syncthreads();
    for (int st = 64; st > 0; st >>= 1) { if (t < st) red[t] += red[t + st]; __syncthreads(); }
    const float mu = red[0] / Dd;
    __syncthreads();

    float v = 0.0f;
    for (int d = t; d < Dd; d += 128) { float dx = xr[d] - mu; v += dx * dx; }
    red[t] = v; __syncthreads();
    for (int st = 64; st > 0; st >>= 1) { if (t < st) red[t] += red[t + st]; __syncthreads(); }
    const float rs = rsqrtf(red[0] / Dd + 1e-5f);

    for (int d = t; d < Dd; d += 128) {
        float xn = (xr[d] - mu) * rs;
        if (o0) o0[ob + d] = (_Float16)(xn * w0[d] + b0[d]);
        if (o1) o1[ob + d] = (_Float16)(xn * w1[d] + b1[d]);
        if (o2) o2[ob + d] = (_Float16)(xn * w2[d] + b2[d]);
    }
}

// ---------------------------------------------------------------------------
// Row softmax over first Nreal cols; pad cols/rows -> 0 (f16 out for WMMA A).
// ---------------------------------------------------------------------------
__global__ __launch_bounds__(128) void softmax_kernel(
    const float* __restrict__ attn, _Float16* __restrict__ probs,
    int Npad, int Nreal, int Mreal)
{
    const int row = blockIdx.x, b = blockIdx.y, t = threadIdx.x;
    __shared__ float red[128];
    const long long base = ((long long)b * gridDim.x + row) * Npad;

    if (row >= Mreal) {
        for (int n = t; n < Npad; n += 128) probs[base + n] = (_Float16)0.0f;
        return;
    }
    float mx = -3.4e38f;
    for (int n = t; n < Nreal; n += 128) mx = fmaxf(mx, attn[base + n]);
    red[t] = mx; __syncthreads();
    for (int st = 64; st > 0; st >>= 1) { if (t < st) red[t] = fmaxf(red[t], red[t + st]); __syncthreads(); }
    mx = red[0]; __syncthreads();

    float s = 0.0f;
    for (int n = t; n < Nreal; n += 128) s += expf(attn[base + n] - mx);
    red[t] = s; __syncthreads();
    for (int st = 64; st > 0; st >>= 1) { if (t < st) red[t] += red[t + st]; __syncthreads(); }
    const float inv = 1.0f / red[0];

    for (int n = t; n < Npad; n += 128)
        probs[base + n] = (_Float16)((n < Nreal) ? expf(attn[base + n] - mx) * inv : 0.0f);
}

// --------------------------- small helper kernels --------------------------
__global__ void cvt_f32_f16(const float* __restrict__ in, _Float16* __restrict__ out, int n) {
    int i = blockIdx.x * 256 + threadIdx.x;
    if (i < n) out[i] = (_Float16)in[i];
}

__global__ __launch_bounds__(256) void parts_base_kernel(
    const float* __restrict__ w2v, const float* __restrict__ W, float* __restrict__ pb)
{
    int d = blockIdx.x * 256 + threadIdx.x;   // < 768
    int l = blockIdx.y;                       // < 312
    float s = 0.0f;
    for (int w = 0; w < W2VD; ++w) s += w2v[l * W2VD + w] * W[w * D + d];
    pb[l * D + d] = s;
}

__global__ __launch_bounds__(256) void rowmax_kernel(const float* __restrict__ pb, float* __restrict__ g) {
    __shared__ float red[256];
    int l = blockIdx.x, t = threadIdx.x;
    float m = -3.4e38f;
    for (int d = t; d < D; d += 256) m = fmaxf(m, pb[l * D + d]);
    red[t] = m; __syncthreads();
    for (int st = 128; st > 0; st >>= 1) { if (t < st) red[t] = fmaxf(red[t], red[t + st]); __syncthreads(); }
    if (t == 0) g[l] = red[0];
}

__global__ __launch_bounds__(320) void sr_kernel(
    const float* __restrict__ g,
    const float* __restrict__ w1, const float* __restrict__ b1,
    const float* __restrict__ w2, const float* __restrict__ b2,
    float* __restrict__ gate)
{
    __shared__ float t1[L];
    int i = threadIdx.x;
    if (i < L) {
        float s = b1[i];
        for (int j = 0; j < L; ++j) s += g[j] * w1[i * L + j];
        t1[i] = gelu_exact(s);
    }
    __syncthreads();
    if (i < L) {
        float s = b2[i];
        for (int j = 0; j < L; ++j) s += t1[j] * w2[i * L + j];
        gate[i] = 1.0f / (1.0f + expf(-s));
    }
}

__global__ __launch_bounds__(256) void parts_scale_kernel(
    const float* __restrict__ pb, const float* __restrict__ gate, float* __restrict__ parts)
{
    int d = blockIdx.x * 256 + threadIdx.x;
    int l = blockIdx.y;  // < LP
    parts[l * D + d] = (l < L) ? (1.0f + gate[l]) * pb[l * D + d] : 0.0f;
}

__global__ __launch_bounds__(256) void mean_kernel(const float* __restrict__ feats, float* __restrict__ outbar) {
    int d = blockIdx.x * 256 + threadIdx.x;  // < 768
    int b = blockIdx.y;
    float s = 0.0f;
    for (int p = 0; p < P; ++p) s += feats[((long long)b * PP + p) * D + d];
    outbar[b * D + d] = s * (1.0f / (float)P);
}

__global__ __launch_bounds__(512) void attnorm_kernel(const float* __restrict__ att, float* __restrict__ attn_n) {
    __shared__ float red[512];
    int c = blockIdx.x, l = threadIdx.x;
    float a = (l < L) ? att[c * L + l] : 0.0f;
    red[l] = a * a; __syncthreads();
    for (int st = 256; st > 0; st >>= 1) { if (l < st) red[l] += red[l + st]; __syncthreads(); }
    float rn = 1.0f / (sqrtf(red[0]) + 1e-5f);
    if (l < L) attn_n[c * L + l] = a * rn;
}

__global__ __launch_bounds__(512) void score_vec_kernel(
    const float* __restrict__ outbar, const float* __restrict__ feats0,
    const float* __restrict__ V, const float* __restrict__ V2,
    float* __restrict__ sVn, float* __restrict__ sV2n)
{
    __shared__ float red[512];
    int b = blockIdx.x, l = threadIdx.x;
    float a = 0.0f, c = 0.0f;
    if (l < L) {
        const float* ob = outbar + (long long)b * D;
        const float* fg = feats0 + (long long)b * 197 * D;   // CLS row
        for (int d = 0; d < D; ++d) {
            a += ob[d] * V[d * L + l];
            c += fg[d] * V2[d * L + l];
        }
    }
    red[l] = a * a; __syncthreads();
    for (int st = 256; st > 0; st >>= 1) { if (l < st) red[l] += red[l + st]; __syncthreads(); }
    float rn1 = 1.0f / (sqrtf(red[0]) + 1e-5f);
    __syncthreads();
    red[l] = c * c; __syncthreads();
    for (int st = 256; st > 0; st >>= 1) { if (l < st) red[l] += red[l + st]; __syncthreads(); }
    float rn2 = 1.0f / (sqrtf(red[0]) + 1e-5f);
    if (l < L) { sVn[b * L + l] = a * rn1; sV2n[b * L + l] = c * rn2; }
}

__global__ void final_kernel(const float* __restrict__ sVn, const float* __restrict__ sV2n,
                             const float* __restrict__ attn_n, float* __restrict__ out)
{
    int idx = blockIdx.x * 256 + threadIdx.x;
    if (idx >= BB * NC) return;
    int b = idx / NC, c = idx % NC;
    float s1 = 0.0f, s2 = 0.0f;
    for (int l = 0; l < L; ++l) {
        float an = attn_n[c * L + l];
        s1 += sVn[b * L + l] * an;
        s2 += sV2n[b * L + l] * an;
    }
    out[idx] = 20.0f * (0.8f * s1 + 0.2f * s2);
}

// ---------------------------------------------------------------------------
extern "C" void kernel_launch(void* const* d_in, const int* in_sizes, int n_in,
                              void* d_out, int out_size, void* d_ws, size_t ws_size,
                              hipStream_t stream)
{
    // -------- inputs (setup_inputs order) --------
    const float* feats_0 = (const float*)d_in[0];
    const float* att_all = (const float*)d_in[1];
    // d_in[2] = seen_att (unused by reference output)
    const float* w2v     = (const float*)d_in[3];
    const float* Wm      = (const float*)d_in[4];
    const float* Vm      = (const float*)d_in[5];
    const float* V2m     = (const float*)d_in[6];
    const float* sr1w    = (const float*)d_in[7];
    const float* sr1b    = (const float*)d_in[8];
    const float* sr2w    = (const float*)d_in[9];
    const float* sr2b    = (const float*)d_in[10];
    const float* mlp_n_w = (const float*)d_in[11];
    const float* mlp_n_b = (const float*)d_in[12];
    const float* mlp1_w  = (const float*)d_in[13];
    const float* mlp1_b  = (const float*)d_in[14];
    const float* mlp2_w  = (const float*)d_in[15];
    const float* mlp2_b  = (const float*)d_in[16];
    const float* e_nq_w = (const float*)d_in[17]; const float* e_nq_b = (const float*)d_in[18];
    const float* e_nk_w = (const float*)d_in[19]; const float* e_nk_b = (const float*)d_in[20];
    const float* e_nv_w = (const float*)d_in[21]; const float* e_nv_b = (const float*)d_in[22];
    const float* e_q_w  = (const float*)d_in[23]; const float* e_q_b  = (const float*)d_in[24];
    const float* e_k_w  = (const float*)d_in[25]; const float* e_k_b  = (const float*)d_in[26];
    const float* e_v_w  = (const float*)d_in[27]; const float* e_v_b  = (const float*)d_in[28];
    const float* e_p_w  = (const float*)d_in[29]; const float* e_p_b  = (const float*)d_in[30];
    const float* d_nq_w = (const float*)d_in[31]; const float* d_nq_b = (const float*)d_in[32];
    const float* d_nk_w = (const float*)d_in[33]; const float* d_nk_b = (const float*)d_in[34];
    const float* d_nv_w = (const float*)d_in[35]; const float* d_nv_b = (const float*)d_in[36];
    const float* d_q_w  = (const float*)d_in[37]; const float* d_q_b  = (const float*)d_in[38];
    const float* d_k_w  = (const float*)d_in[39]; const float* d_k_b  = (const float*)d_in[40];
    const float* d_v_w  = (const float*)d_in[41]; const float* d_v_b  = (const float*)d_in[42];
    const float* d_p_w  = (const float*)d_in[43]; const float* d_p_b  = (const float*)d_in[44];

    // -------- workspace carve (persistent + lifetime-reused big slots) --------
    char*  w   = (char*)d_ws;
    size_t off = 0;
    auto take = [&](size_t bytes) -> char* {
        char* p = w + off;
        off += (bytes + 255) & ~(size_t)255;
        return p;
    };

    const size_t DD2   = (size_t)D * D * 2;
    const size_t HD2   = (size_t)HID * D * 2;
    const size_t ACT_H = (size_t)BB * PP * D * 2;    // 100 MB
    const size_t PRT_H = (size_t)BB * LP * D * 2;    // 126 MB
    const size_t PRT_F = (size_t)BB * LP * D * 4;    // 252 MB
    const size_t ATT_F = (size_t)BB * LP * PP * 4;   //  84 MB
    const size_t MLP_H = (size_t)BB * PP * HID * 2;  // 402 MB

    // persistent small region
    _Float16* wqe = (_Float16*)take(DD2);  _Float16* wke = (_Float16*)take(DD2);
    _Float16* wve = (_Float16*)take(DD2);  _Float16* wpe = (_Float16*)take(DD2);
    _Float16* wqd = (_Float16*)take(DD2);  _Float16* wkd = (_Float16*)take(DD2);
    _Float16* wvd = (_Float16*)take(DD2);  _Float16* wpd = (_Float16*)take(DD2);
    _Float16* w1h = (_Float16*)take(HD2);  _Float16* w2h = (_Float16*)take(HD2);
    float* pb     = (float*)take((size_t)L * D * 4);
    float* gvec   = (float*)take((size_t)L * 4);
    float* gate   = (float*)take((size_t)L * 4);
    float* parts  = (float*)take((size_t)LP * D * 4);
    _Float16* qn_enc_h = (_Float16*)take((size_t)LP * D * 2);
    _Float16* qq_enc_h = (_Float16*)take((size_t)LP * D * 2);
    float* outbar = (float*)take((size_t)BB * D * 4);
    float* sVn    = (float*)take((size_t)BB * L * 4);
    float* sV2n   = (float*)take((size_t)BB * L * 4);
    float* att_n  = (float*)take((size_t)NC * L * 4);

    // big reused slots (lifetimes disjoint within each slot)
    char* S1  = take(ACT_H);   // kn_enc_h -> qn_dec_h -> out_dec_h
    char* S2  = take(ACT_H);   // vn_enc_h -> qq_dec_h
    char* S3  = take(ACT_H);   // lnm_h    -> attn_dec (f32, 84MB <= slot)
    char* S4  = take(ACT_H);   // kk_enc_h -> probs_enc_h -> probs_dec_h
    char* S5  = take(ACT_H);   // vvT_enc_h
    char* S6  = take(ATT_F);   // attn_enc (f32)
    char* S7  = take(PRT_H);   // out_enc_h -> kk_dec_h
    char* S8  = take(PRT_F);   // parts_in(f32) -> feats1(f32) -> vvT_dec_h
    char* S9  = take(PRT_H);   // kn_dec_h
    char* S10 = take(PRT_H);   // vn_dec_h
    char* S11 = take(MLP_H);   // h_mlp_h -> feats(f32)

    _Float16* kn_enc_h  = (_Float16*)S1;  _Float16* qn_dec_h = (_Float16*)S1;
    _Float16* out_dec_h = (_Float16*)S1;
    _Float16* vn_enc_h  = (_Float16*)S2;  _Float16* qq_dec_h = (_Float16*)S2;
    _Float16* lnm_h     = (_Float16*)S3;  float*    attn_dec = (float*)S3;
    _Float16* kk_enc_h  = (_Float16*)S4;  _Float16* probs_e  = (_Float16*)S4;
    _Float16* probs_d   = (_Float16*)S4;
    _Float16* vvT_enc_h = (_Float16*)S5;
    float*    attn_enc  = (float*)S6;
    _Float16* out_enc_h = (_Float16*)S7;  _Float16* kk_dec_h = (_Float16*)S7;
    float*    parts_in  = (float*)S8;     float*    feats1   = (float*)S8;
    _Float16* vvT_dec_h = (_Float16*)S8;
    _Float16* kn_dec_h  = (_Float16*)S9;
    _Float16* vn_dec_h  = (_Float16*)S10;
    _Float16* h_mlp_h   = (_Float16*)S11; float*    feats_f  = (float*)S11;

    const long long PPD = (long long)PP * D, LPD = (long long)LP * D;
    const long long PPH = (long long)PP * HID;
    const long long LPP = (long long)LP * PP, PPL = (long long)PP * LP;

    auto gemm = [&](const _Float16* A, long long sA, const _Float16* Bw, long long sB,
                    const float* bias, const float* resid, long long sR, int ldR, int Mreal,
                    float* oF, long long sOF, _Float16* oH, long long sOH,
                    int Mpad, int N, int K, float alpha, int act, int trans, int batches) {
        dim3 g(N / 64, Mpad / 64, batches);
        wmma_gemm<<<g, 128, 0, stream>>>(A, sA, Bw, sB, bias, resid, sR, ldR, Mreal,
                                         oF, sOF, oH, sOH, N, K, alpha, act, trans);
    };
    auto cvt = [&](const float* in, _Float16* out, int n) {
        cvt_f32_f16<<<(n + 255) / 256, 256, 0, stream>>>(in, out, n);
    };

    // ---------------- weights -> f16 ----------------
    cvt(e_q_w, wqe, D * D);  cvt(e_k_w, wke, D * D);
    cvt(e_v_w, wve, D * D);  cvt(e_p_w, wpe, D * D);
    cvt(d_q_w, wqd, D * D);  cvt(d_k_w, wkd, D * D);
    cvt(d_v_w, wvd, D * D);  cvt(d_p_w, wpd, D * D);
    cvt(mlp1_w, w1h, HID * D);  cvt(mlp2_w, w2h, D * HID);

    // ---------------- batch-independent precompute ----------------
    parts_base_kernel<<<dim3(3, L), 256, 0, stream>>>(w2v, Wm, pb);
    rowmax_kernel<<<L, 256, 0, stream>>>(pb, gvec);
    sr_kernel<<<1, 320, 0, stream>>>(gvec, sr1w, sr1b, sr2w, sr2b, gate);
    parts_scale_kernel<<<dim3(3, LP), 256, 0, stream>>>(pb, gate, parts);
    ln_kernel<<<dim3(LP, 1), 128, 0, stream>>>(parts, 0, D, L, D, LPD,
        e_nq_w, e_nq_b, qn_enc_h, nullptr, nullptr, nullptr, nullptr, nullptr, nullptr);
    gemm(qn_enc_h, 0, wqe, 0, e_q_b, nullptr, 0, 0, 0,
         nullptr, 0, qq_enc_h, 0, LP, D, D, 1.0f, 0, 0, 1);

    // ---------------- per-batch pipeline ----------------
    // LN(feats_in) three ways: enc-k, enc-v, mlp-n  (pad rows -> 0)
    ln_kernel<<<dim3(PP, BB), 128, 0, stream>>>(feats_0 + D, (long long)197 * D, D, P, D, PPD,
        e_nk_w, e_nk_b, kn_enc_h, e_nv_w, e_nv_b, vn_enc_h, mlp_n_w, mlp_n_b, lnm_h);

    gemm(kn_enc_h, PPD, wke, 0, e_k_b, nullptr, 0, 0, 0,
         nullptr, 0, kk_enc_h, PPD, PP, D, D, 1.0f, 0, 0, BB);                  // kk
    gemm(vn_enc_h, PPD, wve, 0, e_v_b, nullptr, 0, 0, 0,
         nullptr, 0, vvT_enc_h, PPD, PP, D, D, 1.0f, 0, 1, BB);                 // vv^T
    gemm(qq_enc_h, 0, kk_enc_h, PPD, nullptr, nullptr, 0, 0, 0,
         attn_enc, LPP, nullptr, 0, LP, PP, D, ATT_SCALE, 1, 0, BB);            // relu(qk)*s
    softmax_kernel<<<dim3(LP, BB), 128, 0, stream>>>(attn_enc, probs_e, PP, P, L);
    gemm(probs_e, LPP, vvT_enc_h, PPD, nullptr, nullptr, 0, 0, 0,
         nullptr, 0, out_enc_h, LPD, LP, D, PP, 1.0f, 0, 0, BB);                // probs@vv
    gemm(out_enc_h, LPD, wpe, 0, e_p_b, parts, 0, D, L,
         parts_in, LPD, nullptr, 0, LP, D, D, 1.0f, 0, 0, BB);                  // parts_in
    ln_kernel<<<dim3(LP, BB), 128, 0, stream>>>(parts_in, LPD, D, L, D, LPD,
        d_nk_w, d_nk_b, kn_dec_h, d_nv_w, d_nv_b, vn_dec_h, nullptr, nullptr, nullptr);

    // MLP branch on feats_in
    gemm(lnm_h, PPD, w1h, 0, mlp1_b, nullptr, 0, 0, 0,
         nullptr, 0, h_mlp_h, PPH, PP, HID, D, 1.0f, 2, 0, BB);                 // gelu(fc1)
    gemm(h_mlp_h, PPH, w2h, 0, mlp2_b, feats_0 + D, (long long)197 * D, D, P,
         feats1, PPD, nullptr, 0, PP, D, HID, 1.0f, 0, 0, BB);                  // feats1
    ln_kernel<<<dim3(PP, BB), 128, 0, stream>>>(feats1, PPD, D, P, D, PPD,
        d_nq_w, d_nq_b, qn_dec_h, nullptr, nullptr, nullptr, nullptr, nullptr, nullptr);

    // decoder attention
    gemm(qn_dec_h, PPD, wqd, 0, d_q_b, nullptr, 0, 0, 0,
         nullptr, 0, qq_dec_h, PPD, PP, D, D, 1.0f, 0, 0, BB);
    gemm(kn_dec_h, LPD, wkd, 0, d_k_b, nullptr, 0, 0, 0,
         nullptr, 0, kk_dec_h, LPD, LP, D, D, 1.0f, 0, 0, BB);
    gemm(vn_dec_h, LPD, wvd, 0, d_v_b, nullptr, 0, 0, 0,
         nullptr, 0, vvT_dec_h, LPD, LP, D, D, 1.0f, 0, 1, BB);
    gemm(qq_dec_h, PPD, kk_dec_h, LPD, nullptr, nullptr, 0, 0, 0,
         attn_dec, PPL, nullptr, 0, PP, LP, D, ATT_SCALE, 1, 0, BB);
    softmax_kernel<<<dim3(PP, BB), 128, 0, stream>>>(attn_dec, probs_d, LP, L, P);
    gemm(probs_d, PPL, vvT_dec_h, LPD, nullptr, nullptr, 0, 0, 0,
         nullptr, 0, out_dec_h, PPD, PP, D, LP, 1.0f, 0, 0, BB);
    gemm(out_dec_h, PPD, wpd, 0, d_p_b, feats_0 + D, (long long)197 * D, D, P,
         feats_f, PPD, nullptr, 0, PP, D, D, 1.0f, 0, 0, BB);                   // feats

    // ---------------- scoring head ----------------
    mean_kernel<<<dim3(3, BB), 256, 0, stream>>>(feats_f, outbar);
    attnorm_kernel<<<NC, 512, 0, stream>>>(att_all, att_n);
    score_vec_kernel<<<BB, 512, 0, stream>>>(outbar, feats_0, Vm, V2m, sVn, sV2n);
    final_kernel<<<(BB * NC + 255) / 256, 256, 0, stream>>>(sVn, sV2n, att_n, (float*)d_out);

    (void)in_sizes; (void)n_in; (void)out_size; (void)ws_size;
}
